// JambaForAudioGeneration_S_In_S_Out_15496242004690
// MI455X (gfx1250) — matrix operand order repaired and checked by reading, not compiled, not run on hardware
//
#include <hip/hip_runtime.h>

// ---------------- model constants ----------------
#define BATCH        4
#define SEQ          1024
#define ROWS         4096          // BATCH*SEQ
#define DMODEL       1024
#define DINNER       2048
#define DSTATE       128
#define NHEADS       32
#define HEADDIM      64
#define CHUNK        64
#define NCHUNK       16
#define CONVDIM      2304          // DINNER + 2*DSTATE
#define DINPROJ      4384          // 2*DINNER + 2*DSTATE + NHEADS
#define NATTNHEADS   16
#define EPS          1e-5f

typedef __bf16 bf16_t;
typedef __attribute__((ext_vector_type(16))) __bf16 v16bf;
typedef __attribute__((ext_vector_type(8)))  float  v8f;

__device__ __forceinline__ bf16_t f2bf(float f) {
  unsigned u = __builtin_bit_cast(unsigned, f);
  unsigned short h = (unsigned short)(u >> 16);
  return __builtin_bit_cast(bf16_t, h);
}

__device__ __forceinline__ v8f wmma_bf16(v16bf a, v16bf b, v8f c) {
  // D = A(16x32) * B(32x16) + C, f32 accumulate
  return __builtin_amdgcn_wmma_f32_16x16x32_bf16(false, a, false, b, (short)0, c,
                                                 false, false);
}

// generic->LDS byte offset (flat LDS aperture keeps offset in addr[31:0])
__device__ __forceinline__ unsigned lds_off(const void* p) {
  return (unsigned)(unsigned long long)p;
}

// ----- fragment builders, bf16 LDS tiles -----
// A fragment: 16x32 tile row-major; lane l: m=l&15, half h=l>>4;
// VGPR0-7: K=h*16..h*16+15
__device__ __forceinline__ v16bf frag_A(const bf16_t* tile, int stride, int m0, int k0) {
  int lane = threadIdx.x & 31;
  int hh = (lane >> 4) & 1;
  const bf16_t* p = tile + (m0 + (lane & 15)) * stride + k0;
  v16bf a;
#pragma unroll
  for (int i = 0; i < 8; ++i) a[i] = p[hh * 16 + i];
#pragma unroll
  for (int i = 0; i < 8; ++i) a[8 + i] = p[hh * 16 + 8 + i];
  return a;
}

// B fragment from transposed tile tileT[n][k]; lane l: n=l&15, K = (l>>4)*16..+15
__device__ __forceinline__ v16bf frag_B(const bf16_t* tileT, int stride, int n0, int k0) {
  int lane = threadIdx.x & 31;
  const bf16_t* p = tileT + (n0 + (lane & 15)) * stride + k0 + ((lane >> 4) & 1) * 16;
  v16bf b;
#pragma unroll
  for (int i = 0; i < 16; ++i) b[i] = p[i];
  return b;
}

// B fragment from row-major K x N tile (strided over k)
__device__ __forceinline__ v16bf frag_B_rm(const bf16_t* tile, int stride, int n0, int k0) {
  int lane = threadIdx.x & 31;
  const bf16_t* p = tile + (k0 + ((lane >> 4) & 1) * 16) * stride + n0 + (lane & 15);
  v16bf b;
#pragma unroll
  for (int i = 0; i < 16; ++i) b[i] = p[i * stride];
  return b;
}

// ----- fragment builders from f32 LDS tiles (convert at read) -----
__device__ __forceinline__ v16bf frag_A_f32(const float* tile, int stride, int m0, int k0) {
  int lane = threadIdx.x & 31;
  int hh = (lane >> 4) & 1;
  const float* p = tile + (m0 + (lane & 15)) * stride + k0;
  v16bf a;
#pragma unroll
  for (int i = 0; i < 8; ++i) a[i] = f2bf(p[hh * 16 + i]);
#pragma unroll
  for (int i = 0; i < 8; ++i) a[8 + i] = f2bf(p[hh * 16 + 8 + i]);
  return a;
}

__device__ __forceinline__ v16bf frag_B_f32(const float* tileT, int stride, int n0, int k0) {
  int lane = threadIdx.x & 31;
  const float* p = tileT + (n0 + (lane & 15)) * stride + k0 + ((lane >> 4) & 1) * 16;
  v16bf b;
#pragma unroll
  for (int i = 0; i < 16; ++i) b[i] = f2bf(p[i]);
  return b;
}

// ---------------- rmsnorm ----------------
__global__ __launch_bounds__(256) void rmsnorm_kernel(const float* __restrict__ X,
                                                      const float* __restrict__ w,
                                                      float* __restrict__ out, int D) {
  int row = blockIdx.x;
  const float* xr = X + (long)row * D;
  float s = 0.f;
  for (int i = threadIdx.x; i < D; i += 256) { float v = xr[i]; s += v * v; }
  __shared__ float red[256];
  red[threadIdx.x] = s;
  __syncthreads();
  for (int st = 128; st > 0; st >>= 1) {
    if (threadIdx.x < st) red[threadIdx.x] += red[threadIdx.x + st];
    __syncthreads();
  }
  float scale = rsqrtf(red[0] / (float)D + EPS);
  for (int i = threadIdx.x; i < D; i += 256)
    out[(long)row * D + i] = xr[i] * scale * w[i];
}

// ---------------- residual add ----------------
__global__ void add_kernel(float* __restrict__ a, const float* __restrict__ b, int n) {
  int i = blockIdx.x * blockDim.x + threadIdx.x;
  if (i < n) a[i] += b[i];
}

// ---------------- WMMA GEMM: C(MxN) = A(MxK) @ W(KxN) ----------------
// 256 threads (8 waves); block tile 128x64, K-step 32.
// Wave w owns rows m0+w*16..+15 and all 4 n-subtiles (4 WMMAs / K-step / wave).
// Software-pipelined: double-buffered LDS, CDNA5 async-to-LDS staging tracked
// on ASYNCcnt. Tile k+1's 12 async ops are issued, then s_wait_asynccnt 12
// drains only tile k's (in-order completion), so global loads overlap WMMA.
// Requires M % 128 == 0, K % 32 == 0 (true at all call sites). Ragged N edge
// handled by clamping the W column (garbage columns are never stored).
__global__ __launch_bounds__(256) void gemm_kernel(const float* __restrict__ A,
                                                   const float* __restrict__ W,
                                                   float* __restrict__ C,
                                                   int M, int N, int K,
                                                   int lda, int ldw, int ldc) {
  __shared__ float sAf[2][128 * 32];   // [m][k]
  __shared__ float sBf[2][64 * 32];    // transposed: [n][k]
  int m0 = blockIdx.y * 128;
  int n0 = blockIdx.x * 64;
  int w  = threadIdx.x >> 5;

  v8f acc[4];
#pragma unroll
  for (int nt = 0; nt < 4; ++nt) acc[nt] = v8f{0, 0, 0, 0, 0, 0, 0, 0};

  // stage one 128x32 A tile + 64x32 W tile (transposed) into LDS buffer `buf`
  auto stage = [&](int k0, int buf) {
    const float* Ablk = A + (long)m0 * lda + k0;
    unsigned ldsA = lds_off(&sAf[buf][0]);
    unsigned ldsB = lds_off(&sBf[buf][0]);
#pragma unroll
    for (int j = 0; j < 4; ++j) {
      int e   = threadIdx.x + j * 256;   // 0..1023 float4 chunks
      int row = e >> 3, q = e & 7;
      const float* ga = Ablk + (long)row * lda + q * 4;
      unsigned la = ldsA + (unsigned)((row * 32 + q * 4) * 4);
      asm volatile("global_load_async_to_lds_b128 %0, %1, off"
                   :: "v"(la), "v"(ga) : "memory");
    }
#pragma unroll
    for (int j = 0; j < 8; ++j) {
      int e  = threadIdx.x + j * 256;    // 0..2047 dwords, transposing scatter
      int kk = e >> 6, nn = e & 63;
      int gn = n0 + nn;
      gn = (gn < N) ? gn : (N - 1);      // branchless clamp for ragged N
      const float* gw = W + (long)(k0 + kk) * ldw + gn;
      unsigned lb = ldsB + (unsigned)((nn * 32 + kk) * 4);
      asm volatile("global_load_async_to_lds_b32 %0, %1, off"
                   :: "v"(lb), "v"(gw) : "memory");
    }
  };

  int KT = K >> 5;
  stage(0, 0);
  for (int kt = 0; kt < KT; ++kt) {
    if (kt + 1 < KT) {
      stage((kt + 1) << 5, (kt + 1) & 1);           // prefetch next tile
      asm volatile("s_wait_asynccnt 12" ::: "memory"); // drain only tile kt
    } else {
      asm volatile("s_wait_asynccnt 0" ::: "memory");
    }
    __syncthreads();

    const float* Af = sAf[kt & 1];
    const float* Bf = sBf[kt & 1];
    v16bf a = frag_A_f32(Af, 32, w * 16, 0);
#pragma unroll
    for (int nt = 0; nt < 4; ++nt)
      acc[nt] = wmma_bf16(a, frag_B_f32(Bf, 32, nt * 16, 0), acc[nt]);
    __syncthreads();   // readers done before this buffer is restaged
  }

  int lane  = threadIdx.x & 31;
  int ln    = lane & 15;
  int hh    = (lane >> 4) & 1;
  int mbase = m0 + w * 16 + hh * 8;
#pragma unroll
  for (int nt = 0; nt < 4; ++nt) {
    int nn = n0 + nt * 16 + ln;
    if (nn < N) {
#pragma unroll
      for (int r = 0; r < 8; ++r)
        C[(long)(mbase + r) * ldc + nn] = acc[nt][r];
    }
  }
}

// ---------------- mamba: dt softplus + A*dt ----------------
__global__ void dt_kernel(const float* __restrict__ zxbcdt,
                          const float* __restrict__ dt_bias,
                          const float* __restrict__ A_log,
                          float* __restrict__ DT, float* __restrict__ DA) {
  int idx = blockIdx.x * blockDim.x + threadIdx.x;   // ROWS*NHEADS
  if (idx >= ROWS * NHEADS) return;
  int r = idx >> 5, h = idx & 31;
  float raw = zxbcdt[(long)r * DINPROJ + (DINPROJ - NHEADS) + h] + dt_bias[h];
  float dt  = (raw > 20.f) ? raw : log1pf(expf(raw));
  float Ah  = -expf(A_log[h]);
  DT[idx] = dt;
  DA[idx] = dt * Ah;
}

// ---------------- causal conv1d + silu ----------------
__global__ void conv_kernel(const float* __restrict__ zxbcdt,
                            const float* __restrict__ cw,
                            const float* __restrict__ cb,
                            float* __restrict__ out) {
  long idx = blockIdx.x * (long)blockDim.x + threadIdx.x;
  if (idx >= (long)ROWS * CONVDIM) return;
  int  cc = (int)(idx % CONVDIM);
  long r  = idx / CONVDIM;
  int  l  = (int)(r & (SEQ - 1));
  float acc = cb[cc];
#pragma unroll
  for (int k = 0; k < 4; ++k) {
    int li = l - 3 + k;
    if (li >= 0)
      acc += cw[cc * 4 + k] * zxbcdt[(r + k - 3) * (long)DINPROJ + DINNER + cc];
  }
  out[idx] = acc / (1.f + expf(-acc));   // silu
}

// ---------------- per-chunk total of dA ----------------
__global__ void acslast_kernel(const float* __restrict__ DA, float* __restrict__ ACSL) {
  int idx = blockIdx.x * blockDim.x + threadIdx.x;   // [b][c][h] = 2048
  if (idx >= BATCH * NCHUNK * NHEADS) return;
  int h = idx & 31, c = (idx >> 5) & 15, b = idx >> 9;
  float s = 0.f;
  long base = (long)(b * SEQ + c * CHUNK);
  for (int i = 0; i < CHUNK; ++i) s += DA[(base + i) * NHEADS + h];
  ACSL[idx] = s;
}

// ---------------- SSD chunk pass 1: Y_diag + chunk states ----------------
__global__ __launch_bounds__(256) void ssd_chunk1(const float* __restrict__ xbc,
                                                  const float* __restrict__ DT,
                                                  const float* __restrict__ DA,
                                                  const float* __restrict__ Dp,
                                                  float* __restrict__ Y,
                                                  float* __restrict__ ST) {
  int g = blockIdx.x;                       // [b][c][h]
  int h = g & 31, c = (g >> 5) & 15, b = g >> 9;
  long rowbase = (long)(b * SEQ + c * CHUNK);

  __shared__ bf16_t sCm[64 * 128];          // [l][n]
  __shared__ bf16_t sBm[64 * 128];          // [s][n]
  __shared__ bf16_t sXdT[64 * 64];          // [p][l]  xd transposed
  __shared__ bf16_t sXdd[64 * 64];          // [p][l]  xd * decay_states
  __shared__ bf16_t sM[64 * 64];            // masked L*G
  __shared__ float  sAcs[64];
  __shared__ float  sdt[64];

  if (threadIdx.x == 0) {
    float s = 0.f;
    for (int i = 0; i < 64; ++i) { s += DA[(rowbase + i) * NHEADS + h]; sAcs[i] = s; }
  }
  if (threadIdx.x < 64) sdt[threadIdx.x] = DT[(rowbase + threadIdx.x) * NHEADS + h];
  __syncthreads();
  float acsLast = sAcs[63];

  for (int e = threadIdx.x; e < 64 * 128; e += 256) {
    int l = e >> 7, n = e & 127;
    long rr = (rowbase + l) * CONVDIM;
    sBm[e] = f2bf(xbc[rr + DINNER + n]);
    sCm[e] = f2bf(xbc[rr + DINNER + DSTATE + n]);
  }
  for (int e = threadIdx.x; e < 64 * 64; e += 256) {
    int l = e >> 6, p = e & 63;
    float xd = xbc[(rowbase + l) * CONVDIM + h * HEADDIM + p] * sdt[l];
    sXdT[p * 64 + l] = f2bf(xd);
    sXdd[p * 64 + l] = f2bf(xd * expf(acsLast - sAcs[l]));
  }
  __syncthreads();

  int w    = threadIdx.x >> 5;
  int lane = threadIdx.x & 31;
  int ln   = lane & 15;
  int hh   = (lane >> 4) & 1;

  // G = Cm(64x128) @ Bm^T -> masked by exp(segsum) -> sM
  for (int t = w; t < 16; t += 8) {
    int mt = (t >> 2) * 16, nt = (t & 3) * 16;
    v8f acc = {0, 0, 0, 0, 0, 0, 0, 0};
#pragma unroll
    for (int ks = 0; ks < 4; ++ks)
      acc = wmma_bf16(frag_A(sCm, 128, mt, ks * 32), frag_B(sBm, 128, nt, ks * 32), acc);
    int ss = nt + ln;
#pragma unroll
    for (int r = 0; r < 8; ++r) {
      int ll = mt + hh * 8 + r;
      float v = (ss <= ll) ? acc[r] * expf(sAcs[ll] - sAcs[ss]) : 0.f;
      sM[ll * 64 + ss] = f2bf(v);
    }
  }
  __syncthreads();

  // Y_diag = sM(64x64) @ xd(64x64) ; plus D[h]*x skip term
  for (int t = w; t < 16; t += 8) {
    int mt = (t >> 2) * 16, nt = (t & 3) * 16;
    v8f acc = {0, 0, 0, 0, 0, 0, 0, 0};
#pragma unroll
    for (int ks = 0; ks < 2; ++ks)
      acc = wmma_bf16(frag_A(sM, 64, mt, ks * 32), frag_B(sXdT, 64, nt, ks * 32), acc);
    int pp = nt + ln;
#pragma unroll
    for (int r = 0; r < 8; ++r) {
      int ll = mt + hh * 8 + r;
      long row = rowbase + ll;
      float xs = xbc[row * CONVDIM + h * HEADDIM + pp];
      Y[row * DINNER + h * HEADDIM + pp] = acc[r] + Dp[h] * xs;
    }
  }

  // states(64p x 128n) = (xd*decay)^T @ Bm
  long stbase = (((long)(b * NCHUNK + c) * NHEADS + h) * HEADDIM) * DSTATE;
  for (int t = w; t < 32; t += 8) {
    int pt = (t >> 3) * 16, ntl = (t & 7) * 16;
    v8f acc = {0, 0, 0, 0, 0, 0, 0, 0};
#pragma unroll
    for (int ks = 0; ks < 2; ++ks)
      acc = wmma_bf16(frag_A(sXdd, 64, pt, ks * 32), frag_B_rm(sBm, 128, ntl, ks * 32), acc);
    int nn = ntl + ln;
#pragma unroll
    for (int r = 0; r < 8; ++r) {
      int pp = pt + hh * 8 + r;
      ST[stbase + (long)pp * DSTATE + nn] = acc[r];
    }
  }
}

// ---------------- inter-chunk state scan (in place) ----------------
__global__ void scan_kernel(float* __restrict__ ST, const float* __restrict__ ACSL) {
  long idx = blockIdx.x * (long)blockDim.x + threadIdx.x;  // b*h*p*n
  if (idx >= (long)BATCH * NHEADS * HEADDIM * DSTATE) return;
  int n = (int)(idx & 127);
  int p = (int)((idx >> 7) & 63);
  int h = (int)((idx >> 13) & 31);
  int b = (int)(idx >> 18);
  float R = 0.f;
  for (int c = 0; c < NCHUNK; ++c) {
    long si = ((((long)(b * NCHUNK + c) * NHEADS + h) * HEADDIM + p) * DSTATE) + n;
    float s = ST[si];
    ST[si] = R;                        // state entering this chunk
    R = R * expf(ACSL[(b * NCHUNK + c) * NHEADS + h]) + s;
  }
}

// ---------------- SSD chunk pass 2: Y += exp(Acs)*Cm@state^T ----------------
__global__ __launch_bounds__(256) void ssd_chunk2(const float* __restrict__ xbc,
                                                  const float* __restrict__ DA,
                                                  const float* __restrict__ ST,
                                                  float* __restrict__ Y) {
  int g = blockIdx.x;
  int h = g & 31, c = (g >> 5) & 15, b = g >> 9;
  long rowbase = (long)(b * SEQ + c * CHUNK);

  __shared__ bf16_t sCm[64 * 128];
  __shared__ bf16_t sSp[64 * 128];          // [p][n]
  __shared__ float  sAcs[64];

  if (threadIdx.x == 0) {
    float s = 0.f;
    for (int i = 0; i < 64; ++i) { s += DA[(rowbase + i) * NHEADS + h]; sAcs[i] = s; }
  }
  long stbase = (((long)(b * NCHUNK + c) * NHEADS + h) * HEADDIM) * DSTATE;
  for (int e = threadIdx.x; e < 64 * 128; e += 256) {
    int l = e >> 7, n = e & 127;
    sCm[e] = f2bf(xbc[(rowbase + l) * CONVDIM + DINNER + DSTATE + n]);
    sSp[e] = f2bf(ST[stbase + e]);
  }
  __syncthreads();

  int w = threadIdx.x >> 5, lane = threadIdx.x & 31;
  int ln = lane & 15, hh = (lane >> 4) & 1;
  for (int t = w; t < 16; t += 8) {
    int mt = (t >> 2) * 16, pt = (t & 3) * 16;
    v8f acc = {0, 0, 0, 0, 0, 0, 0, 0};
#pragma unroll
    for (int ks = 0; ks < 4; ++ks)
      acc = wmma_bf16(frag_A(sCm, 128, mt, ks * 32), frag_B(sSp, 128, pt, ks * 32), acc);
    int pp = pt + ln;
#pragma unroll
    for (int r = 0; r < 8; ++r) {
      int ll = mt + hh * 8 + r;
      long row = rowbase + ll;
      Y[row * DINNER + h * HEADDIM + pp] += expf(sAcs[ll]) * acc[r];
    }
  }
}

// ---------------- gated RMS norm over DINNER (in place on Y) ----------------
__global__ __launch_bounds__(256) void gate_norm_kernel(float* __restrict__ Y,
                                                        const float* __restrict__ zxbcdt,
                                                        const float* __restrict__ w) {
  int row = blockIdx.x;
  float vals[8];
  float s = 0.f;
#pragma unroll
  for (int j = 0; j < 8; ++j) {
    int d  = threadIdx.x + j * 256;
    float z = zxbcdt[(long)row * DINPROJ + d];
    float t = Y[(long)row * DINNER + d] * (z / (1.f + expf(-z)));
    vals[j] = t;
    s += t * t;
  }
  __shared__ float red[256];
  red[threadIdx.x] = s;
  __syncthreads();
  for (int st = 128; st > 0; st >>= 1) {
    if (threadIdx.x < st) red[threadIdx.x] += red[threadIdx.x + st];
    __syncthreads();
  }
  float scale = rsqrtf(red[0] / (float)DINNER + EPS);
#pragma unroll
  for (int j = 0; j < 8; ++j) {
    int d = threadIdx.x + j * 256;
    Y[(long)row * DINNER + d] = vals[j] * scale * w[d];
  }
}

// ---------------- causal flash attention (1 wave / 16 query rows) ----------------
__global__ __launch_bounds__(32) void attn_kernel(const float* __restrict__ Q,
                                                  const float* __restrict__ K,
                                                  const float* __restrict__ V,
                                                  float* __restrict__ O) {
  int g    = blockIdx.x;                 // [b][head][qtile]
  int qt   = g & 63;
  int head = (g >> 6) & 15;
  int b    = g >> 10;
  int q0   = qt * 16;
  int lane = threadIdx.x;
  int ln   = lane & 15;
  int hh   = lane >> 4;

  __shared__ bf16_t sQ[16 * 64];
  __shared__ bf16_t sK[32 * 64];
  __shared__ bf16_t sV[32 * 64];         // row-major [kv][d]
  __shared__ bf16_t sP[16 * 32];

  for (int e = lane; e < 16 * 64; e += 32) {
    int rr = e >> 6, dd = e & 63;
    sQ[e] = f2bf(Q[((long)(b * SEQ + q0 + rr)) * DMODEL + head * 64 + dd] * 0.125f);
  }
  float mI[8], lI[8];
  v8f Oacc[4];
#pragma unroll
  for (int r = 0; r < 8; ++r) { mI[r] = -3e30f; lI[r] = 0.f; }
#pragma unroll
  for (int dt = 0; dt < 4; ++dt) Oacc[dt] = v8f{0, 0, 0, 0, 0, 0, 0, 0};
  __syncthreads();

  int ktmax = (q0 + 15) / 32;
  for (int kt = 0; kt <= ktmax; ++kt) {
    int kv0 = kt * 32;
    for (int e = lane; e < 32 * 64; e += 32) {
      int kv = e >> 6, dd = e & 63;
      long row = (long)(b * SEQ + kv0 + kv) * DMODEL + head * 64 + dd;
      sK[e] = f2bf(K[row]);
      sV[e] = f2bf(V[row]);
    }
    __syncthreads();

    v8f S0 = {0, 0, 0, 0, 0, 0, 0, 0}, S1 = {0, 0, 0, 0, 0, 0, 0, 0};
#pragma unroll
    for (int ks = 0; ks < 2; ++ks) {
      v16bf a = frag_A(sQ, 64, 0, ks * 32);
      S0 = wmma_bf16(a, frag_B(sK, 64, 0, ks * 32), S0);
      S1 = wmma_bf16(a, frag_B(sK, 64, 16, ks * 32), S1);
    }
    float rowmax[8];
#pragma unroll
    for (int r = 0; r < 8; ++r) {
      int qq = q0 + hh * 8 + r;
      float s0 = (kv0 + ln <= qq) ? S0[r] : -3e30f;
      float s1 = (kv0 + 16 + ln <= qq) ? S1[r] : -3e30f;
      S0[r] = s0; S1[r] = s1;
      rowmax[r] = fmaxf(s0, s1);
    }
#pragma unroll
    for (int off = 1; off < 16; off <<= 1)
#pragma unroll
      for (int r = 0; r < 8; ++r)
        rowmax[r] = fmaxf(rowmax[r], __shfl_xor(rowmax[r], off, 32));

    float mNew[8], alpha[8], rowsum[8];
#pragma unroll
    for (int r = 0; r < 8; ++r) {
      mNew[r]  = fmaxf(mI[r], rowmax[r]);
      alpha[r] = expf(mI[r] - mNew[r]);
      float p0 = expf(S0[r] - mNew[r]);
      float p1 = expf(S1[r] - mNew[r]);
      sP[(hh * 8 + r) * 32 + ln]      = f2bf(p0);
      sP[(hh * 8 + r) * 32 + 16 + ln] = f2bf(p1);
      rowsum[r] = p0 + p1;
    }
#pragma unroll
    for (int off = 1; off < 16; off <<= 1)
#pragma unroll
      for (int r = 0; r < 8; ++r)
        rowsum[r] += __shfl_xor(rowsum[r], off, 32);
#pragma unroll
    for (int r = 0; r < 8; ++r) {
      lI[r] = lI[r] * alpha[r] + rowsum[r];
      mI[r] = mNew[r];
    }
#pragma unroll
    for (int dt = 0; dt < 4; ++dt)
#pragma unroll
      for (int r = 0; r < 8; ++r) Oacc[dt][r] *= alpha[r];
    __syncthreads();

    v16bf pa = frag_A(sP, 32, 0, 0);
#pragma unroll
    for (int dt = 0; dt < 4; ++dt)
      Oacc[dt] = wmma_bf16(pa, frag_B_rm(sV, 64, dt * 16, 0), Oacc[dt]);
    __syncthreads();
  }
#pragma unroll
  for (int dt = 0; dt < 4; ++dt)
#pragma unroll
    for (int r = 0; r < 8; ++r) {
      int qq = q0 + hh * 8 + r;
      O[((long)(b * SEQ + qq)) * DMODEL + head * 64 + dt * 16 + ln] =
          Oacc[dt][r] / lI[r];
    }
}

// ---------------- orchestration ----------------
extern "C" void kernel_launch(void* const* d_in, const int* in_sizes, int n_in,
                              void* d_out, int out_size, void* d_ws, size_t ws_size,
                              hipStream_t stream) {
  (void)in_sizes; (void)n_in; (void)out_size; (void)ws_size;
  float* ws = (float*)d_ws;

  const long SZ_X  = (long)ROWS * DMODEL;          // 4,194,304
  const long SZ_B1 = (long)ROWS * DINPROJ;         // 17,956,864
  const long SZ_B2 = (long)ROWS * CONVDIM;         // 9,437,184
  const long SZ_Y  = (long)ROWS * DINNER;          // 8,388,608
  const long SZ_DT = (long)ROWS * NHEADS;          // 131,072
  const long SZ_AL = (long)BATCH * NCHUNK * NHEADS;

  float* X    = ws;
  float* H    = X + SZ_X;
  float* B1   = H + SZ_X;
  float* B2   = B1 + SZ_B1;
  float* Yb   = B2 + SZ_B2;
  float* DTb  = Yb + SZ_Y;
  float* DAb  = DTb + SZ_DT;
  float* ACSL = DAb + SZ_DT;
  float* STb  = ACSL + SZ_AL;

  hipMemcpyAsync(X, d_in[0], SZ_X * sizeof(float), hipMemcpyDeviceToDevice, stream);

  int idx = 1;
  for (int layer = 0; layer < 8; ++layer) {
    const float* norm_w = (const float*)d_in[idx++];
    rmsnorm_kernel<<<ROWS, 256, 0, stream>>>(X, norm_w, H, DMODEL);

    if (layer & 1) {  // attention mixer
      const float* wq = (const float*)d_in[idx++];
      const float* wk = (const float*)d_in[idx++];
      const float* wv = (const float*)d_in[idx++];
      const float* wo = (const float*)d_in[idx++];
      float* Qp = B1;
      float* Kp = B1 + SZ_X;
      float* Vp = B1 + 2 * SZ_X;
      dim3 gg(DMODEL / 64, ROWS / 128);
      gemm_kernel<<<gg, 256, 0, stream>>>(H, wq, Qp, ROWS, DMODEL, DMODEL, DMODEL, DMODEL, DMODEL);
      gemm_kernel<<<gg, 256, 0, stream>>>(H, wk, Kp, ROWS, DMODEL, DMODEL, DMODEL, DMODEL, DMODEL);
      gemm_kernel<<<gg, 256, 0, stream>>>(H, wv, Vp, ROWS, DMODEL, DMODEL, DMODEL, DMODEL, DMODEL);
      attn_kernel<<<BATCH * NATTNHEADS * (SEQ / 16), 32, 0, stream>>>(Qp, Kp, Vp, B2);
      gemm_kernel<<<gg, 256, 0, stream>>>(B2, wo, H, ROWS, DMODEL, DMODEL, DMODEL, DMODEL, DMODEL);
      add_kernel<<<(int)(SZ_X / 256), 256, 0, stream>>>(X, H, (int)SZ_X);
    } else {          // mamba2 mixer
      const float* in_proj    = (const float*)d_in[idx++];
      const float* conv_w     = (const float*)d_in[idx++];
      const float* conv_b     = (const float*)d_in[idx++];
      const float* dt_bias    = (const float*)d_in[idx++];
      const float* A_log      = (const float*)d_in[idx++];
      const float* Dp         = (const float*)d_in[idx++];
      const float* ssm_norm_w = (const float*)d_in[idx++];
      const float* out_proj   = (const float*)d_in[idx++];

      gemm_kernel<<<dim3((DINPROJ + 63) / 64, ROWS / 128), 256, 0, stream>>>(
          H, in_proj, B1, ROWS, DINPROJ, DMODEL, DMODEL, DINPROJ, DINPROJ);
      dt_kernel<<<(ROWS * NHEADS + 255) / 256, 256, 0, stream>>>(B1, dt_bias, A_log, DTb, DAb);
      conv_kernel<<<(int)(((long)ROWS * CONVDIM + 255) / 256), 256, 0, stream>>>(
          B1, conv_w, conv_b, B2);
      acslast_kernel<<<(int)((SZ_AL + 255) / 256), 256, 0, stream>>>(DAb, ACSL);
      ssd_chunk1<<<BATCH * NCHUNK * NHEADS, 256, 0, stream>>>(B2, DTb, DAb, Dp, Yb, STb);
      scan_kernel<<<(int)(((long)BATCH * NHEADS * HEADDIM * DSTATE) / 256), 256, 0, stream>>>(
          STb, ACSL);
      ssd_chunk2<<<BATCH * NCHUNK * NHEADS, 256, 0, stream>>>(B2, DAb, STb, Yb);
      gate_norm_kernel<<<ROWS, 256, 0, stream>>>(Yb, B1, ssm_norm_w);
      gemm_kernel<<<dim3(DMODEL / 64, ROWS / 128), 256, 0, stream>>>(
          Yb, out_proj, H, ROWS, DMODEL, DINNER, DINNER, DMODEL, DMODEL);
      add_kernel<<<(int)(SZ_X / 256), 256, 0, stream>>>(X, H, (int)SZ_X);
    }
  }
  const float* norm_f_w = (const float*)d_in[idx++];
  rmsnorm_kernel<<<ROWS, 256, 0, stream>>>(X, norm_f_w, (float*)d_out, DMODEL);
}
